// DynamicConv2d_29042568855771
// MI455X (gfx1250) — matrix operand order repaired and checked by reading, not compiled, not run on hardware
//
#include <hip/hip_runtime.h>

typedef __attribute__((ext_vector_type(16))) __bf16 v16bf;
typedef __attribute__((ext_vector_type(8)))  float  v8f;
typedef __attribute__((ext_vector_type(4)))  unsigned int v4u;
typedef __attribute__((ext_vector_type(8)))  int v8i;
typedef __attribute__((ext_vector_type(4)))  int v4i;

#define BB 8
#define CC 64
#define HH 128
#define WW 128
#define OO 64
#define CK 576              // C * 3 * 3
#define RR 16               // C / RED
#define NPIX (HH * WW)
#define XW 132              // halo row: [z z x0..x127 z z], valid starts at col 2

#define LDS_A_BYTES (OO * CK * 2)          // 73728
#define LDS_X_BYTES (3 * CC * XW * 2)      // 50688
#define LDS_T_BYTES (CK * 4)               // 2304

// Use the TDM (tensor data mover) to stage A. Header presence selects arity.
#if __has_include(<hip/amd_detail/amd_gfx1250_TDM.h>)
#define TDM_SIX_ARGS 1
#endif

// ---------------------------------------------------------------------------
// Kernel 1: global average pool  ->  pooled[b][c]  (B*C blocks)
// ---------------------------------------------------------------------------
__global__ __launch_bounds__(256) void pool_kernel(const float* __restrict__ x,
                                                   float* __restrict__ pooled) {
  __shared__ float red[256];
  const int bc = blockIdx.x;
  const float* src = x + (size_t)bc * NPIX;
  float s = 0.f;
  for (int i = threadIdx.x; i < NPIX; i += 256) s += src[i];
  red[threadIdx.x] = s;
  __syncthreads();
  for (int off = 128; off > 0; off >>= 1) {
    if ((int)threadIdx.x < off) red[threadIdx.x] += red[threadIdx.x + off];
    __syncthreads();
  }
  if (threadIdx.x == 0) pooled[bc] = red[0] * (1.0f / NPIX);
}

// ---------------------------------------------------------------------------
// Kernel 2: hid = relu(pooled @ w1^T + b1); wdyn = hid @ w2^T + b2  (bf16 out)
// ---------------------------------------------------------------------------
__global__ __launch_bounds__(256) void wdyn_kernel(const float* __restrict__ pooled,
                                                   const float* __restrict__ w1,
                                                   const float* __restrict__ b1,
                                                   const float* __restrict__ w2,
                                                   const float* __restrict__ b2,
                                                   __bf16* __restrict__ wdyn) {
  __shared__ float hid[RR];
  const int blocksPerBatch = (OO * CK) / 256;      // 144
  const int b     = blockIdx.x / blocksPerBatch;
  const int ebase = (blockIdx.x % blocksPerBatch) * 256;

  if (threadIdx.x < RR) {
    float a = b1[threadIdx.x];
    const float* p  = pooled + b * CC;
    const float* wr = w1 + threadIdx.x * CC;
    #pragma unroll 8
    for (int c = 0; c < CC; ++c) a += p[c] * wr[c];
    hid[threadIdx.x] = a > 0.f ? a : 0.f;
  }
  __syncthreads();

  const int e = ebase + threadIdx.x;
  float acc = b2[e];
  const float* wr = w2 + (size_t)e * RR;
  #pragma unroll
  for (int r = 0; r < RR; ++r) acc += hid[r] * wr[r];
  wdyn[(size_t)b * (OO * CK) + e] = (__bf16)acc;
}

// ---------------------------------------------------------------------------
// Kernel 3: out[b,:, y, :] = wdyn[b] (64x576) x im2col patches (576x128)
// One block per (y, b). 8 waves; wave w owns n-tile w (16 cols) x all 4 m-tiles.
// ---------------------------------------------------------------------------
__global__ __launch_bounds__(256) void conv_kernel(const float* __restrict__ x,
                                                   const __bf16* __restrict__ wdyn,
                                                   float* __restrict__ out) {
  extern __shared__ char smem[];
  __bf16* ldsA = (__bf16*)smem;                              // [64][576]
  __bf16* ldsX = (__bf16*)(smem + LDS_A_BYTES);              // [3][64][XW]
  int*    tbl  = (int*)(smem + LDS_A_BYTES + LDS_X_BYTES);   // [576] byte offsets

  const int y   = blockIdx.x;
  const int b   = blockIdx.y;
  const int tid = threadIdx.x;
  const int wave = tid >> 5;

  // ---- Phase 1: stage A via Tensor Data Mover (wave 0 only) ----
  if (wave == 0) {
    const unsigned long long ga =
        (unsigned long long)(uintptr_t)(wdyn + (size_t)b * (OO * CK));
    const unsigned int ldsAoff = (unsigned int)(uintptr_t)ldsA;

    v4u g0;
    g0[0] = 1u;                                    // count=1, user mode
    g0[1] = ldsAoff;                               // lds_addr
    g0[2] = (unsigned int)(ga & 0xffffffffull);    // global_addr[31:0]
    g0[3] = (unsigned int)((ga >> 32) & 0x01ffffffull) | 0x80000000u; // [56:32] | type=2

    v8i g1;
    g1[0] = (int)(1u << 16);        // workgroup_mask=0, data_size=1 (2 bytes)
    g1[1] = (int)(576u << 16);      // tensor_dim0[15:0] = 576
    g1[2] = (int)(64u << 16);       // tensor_dim1[15:0] = 64
    g1[3] = (int)(576u << 16);      // tile_dim0 = 576
    g1[4] = 64;                     // tile_dim1 = 64, tile_dim2 = 0
    g1[5] = 576;                    // tensor_dim0_stride = 576
    g1[6] = 0;
    g1[7] = 0;

    v4i g2 = {0, 0, 0, 0};
    v4i g3 = {0, 0, 0, 0};
#if defined(TDM_SIX_ARGS)
    v8i g4 = {0, 0, 0, 0, 0, 0, 0, 0};
    __builtin_amdgcn_tensor_load_to_lds(g0, g1, g2, g3, g4, 0);
#else
    __builtin_amdgcn_tensor_load_to_lds(g0, g1, g2, g3, 0);
#endif
    __builtin_amdgcn_s_wait_tensorcnt(0);
  }

  // ---- Phase 2: stage x rows y-1..y+1 (f32 -> bf16, pair-packed stores) ----
  if (tid < 3 * CC) {
    const int r  = tid >> 6;          // 0..2  (kh)
    const int c  = tid & 63;
    const int yy = y + r - 1;
    __bf16* row = ldsX + (r * CC + c) * XW;
    union Pk { __bf16 h[2]; unsigned int u; };
    Pk z; z.h[0] = (__bf16)0.f; z.h[1] = (__bf16)0.f;
    ((unsigned int*)row)[0] = z.u;                 // cols 0,1
    ((unsigned int*)(row + 130))[0] = z.u;         // cols 130,131
    unsigned int* rowp = (unsigned int*)(row + 2); // 4-byte aligned
    if ((unsigned)yy < HH) {
      const float4* src = (const float4*)(x + (((size_t)b * CC + c) * HH + yy) * WW);
      #pragma unroll 4
      for (int xx = 0; xx < WW; xx += 4) {
        float4 v = src[xx >> 2];
        Pk a, d;
        a.h[0] = (__bf16)v.x; a.h[1] = (__bf16)v.y;
        d.h[0] = (__bf16)v.z; d.h[1] = (__bf16)v.w;
        rowp[(xx >> 1) + 0] = a.u;
        rowp[(xx >> 1) + 1] = d.u;
      }
    } else {
      #pragma unroll 8
      for (int xx = 0; xx < WW; xx += 2) rowp[xx >> 1] = z.u;
    }
  }

  // ---- Phase 2b: im2col byte-offset table: k -> (c, kh, kw) ----
  for (int k = tid; k < CK; k += 256) {
    const int c  = k / 9;
    const int j  = k - c * 9;
    const int kh = (j * 11) >> 5;    // j / 3
    const int kw = j - kh * 3;
    tbl[k] = ((kh * CC + c) * XW + (kw + 1)) * 2;  // + n*2 at use time
  }
  __syncthreads();

  // ---- Phase 3: WMMA GEMM ----
  const int lane = tid & 31;
  const int lhi  = lane >> 4;         // 0|1: K-half carried by this lane
  const int llo  = lane & 15;
  const int nn   = wave * 16 + llo;   // this lane's output column
  const int nbyte = nn * 2;

  v8f acc[4] = {};

  union AF { v16bf v; uint4 q[2]; };
  union BF { v16bf v; __bf16 e[16]; };

  const __bf16* aBase = ldsA + llo * CK + (lhi << 3);
  const char*   xBase = (const char*)ldsX;

  for (int kc = 0; kc < CK / 32; ++kc) {
    const int kbase = kc * 32;

    // A fragments for all 4 m-tiles: shared address, immediate m-tile offsets
    AF af[4];
    {
      const __bf16* ap = aBase + kbase;
      #pragma unroll
      for (int mt = 0; mt < 4; ++mt) {
        af[mt].q[0] = *(const uint4*)(ap + mt * 16 * CK);        // K k0..k0+7
        af[mt].q[1] = *(const uint4*)(ap + mt * 16 * CK + 16);   // K k0+16..k0+23
      }
    }

    // B fragment: table entries k0..k0+7, k0+16..k0+23 + im2col gathers
    BF bfr;
    {
      const int kb = kbase + (lhi << 3);
      const int4* tp = (const int4*)(tbl + kb);    // 32B aligned
      int4 t0 = tp[0], t1 = tp[1], t2 = tp[4], t3 = tp[5];
      int ofs[16] = {t0.x, t0.y, t0.z, t0.w, t1.x, t1.y, t1.z, t1.w,
                     t2.x, t2.y, t2.z, t2.w, t3.x, t3.y, t3.z, t3.w};
      #pragma unroll
      for (int i = 0; i < 16; ++i)
        bfr.e[i] = *(const __bf16*)(xBase + ofs[i] + nbyte);
    }

    #pragma unroll
    for (int mt = 0; mt < 4; ++mt)
      acc[mt] = __builtin_amdgcn_wmma_f32_16x16x32_bf16(
          false, af[mt].v, false, bfr.v, (short)0, acc[mt], false, false);
  }

  // ---- Phase 4: store D (VGPR r -> M = mt*16 + r + 8*lhi, N = nn) ----
  #pragma unroll
  for (int mt = 0; mt < 4; ++mt) {
    const int mrow = mt * 16 + (lhi << 3);
    float* op = out + (((size_t)b * OO + mrow) * HH + y) * WW + nn;
    #pragma unroll
    for (int r = 0; r < 8; ++r)
      op[(size_t)r * (HH * WW)] = acc[mt][r];
  }
}

// ---------------------------------------------------------------------------
extern "C" void kernel_launch(void* const* d_in, const int* in_sizes, int n_in,
                              void* d_out, int out_size, void* d_ws, size_t ws_size,
                              hipStream_t stream) {
  const float* x  = (const float*)d_in[0];
  const float* w1 = (const float*)d_in[1];
  const float* b1 = (const float*)d_in[2];
  const float* w2 = (const float*)d_in[3];
  const float* b2 = (const float*)d_in[4];
  float* out = (float*)d_out;

  // workspace: [0,2048) pooled f32 (8*64), then wdyn bf16 (8*36864)
  float*  pooled = (float*)d_ws;
  __bf16* wdyn   = (__bf16*)((char*)d_ws + 2048);

  pool_kernel<<<BB * CC, 256, 0, stream>>>(x, pooled);
  wdyn_kernel<<<BB * (OO * CK) / 256, 256, 0, stream>>>(pooled, w1, b1, w2, b2, wdyn);

  const size_t lds = LDS_A_BYTES + LDS_X_BYTES + LDS_T_BYTES;  // ~124 KB
  conv_kernel<<<dim3(HH, BB), 256, lds, stream>>>(x, wdyn, out);
}
